// STGT_64072322122337
// MI455X (gfx1250) — compile-verified, matrix operand order checked by reading, not run on hardware
//
#include <hip/hip_runtime.h>
#include <hip/hip_bf16.h>

// ---------------- constants ----------------
#define NE     4096
#define NT     48
#define ND     128
#define NH     8
#define DH     16
#define KNBR   16
#define KLAP   16
#define NL     2
#define DFF_T  256
#define DFF_S  512
#define MROWS  ((size_t)NE * NT)     // 196608

typedef __bf16 bf16;
typedef __attribute__((ext_vector_type(16))) __bf16 v16bf;
typedef __attribute__((ext_vector_type(8)))  __bf16 v8bf;
typedef __attribute__((ext_vector_type(8)))  float  v8f;
typedef __attribute__((ext_vector_type(4)))  unsigned int u32x4;
typedef __attribute__((ext_vector_type(8)))  int i32x8;
typedef __attribute__((ext_vector_type(4)))  int i32x4;

union FragBF { v16bf v; v8bf h[2]; };

// ---------------- f32 -> bf16 convert ----------------
__global__ void __launch_bounds__(256) f2b_kernel(const float* __restrict__ src,
                                                  bf16* __restrict__ dst, int n) {
  int i = blockIdx.x * 256 + threadIdx.x;
  if (i < n) dst[i] = (bf16)src[i];
}

// ---------------- embedding: t2v + input proj + laplacian PE ----------------
__global__ void __launch_bounds__(256) embed_kernel(
    const float* __restrict__ x, const float* __restrict__ hours,
    const float* __restrict__ U,
    const float* __restrict__ wl, const float* __restrict__ bl,
    const float* __restrict__ wp, const float* __restrict__ bp,
    const float* __restrict__ w_in, const float* __restrict__ b_in,
    const float* __restrict__ w_pe,
    float* __restrict__ hF, bf16* __restrict__ hB) {
  size_t idx = (size_t)blockIdx.x * 256 + threadIdx.x;   // over MROWS*ND
  size_t row = idx >> 7;
  int d = (int)(idx & 127);
  int e = (int)(row / NT);
  float hv = hours[row];
  float in7[7];
  in7[0] = x[row];
  in7[1] = hv * wl[0] + bl[0];
#pragma unroll
  for (int j = 0; j < 5; ++j) in7[2 + j] = __sinf(hv * wp[j] + bp[j]);
  float acc = b_in[d];
#pragma unroll
  for (int j = 0; j < 7; ++j) acc += in7[j] * w_in[d * 7 + j];
  const float* u = U + (size_t)e * KLAP;
#pragma unroll
  for (int k = 0; k < KLAP; ++k) acc += u[k] * w_pe[d * KLAP + k];
  hF[idx] = acc;
  hB[idx] = (bf16)acc;
}

// ---------------- WMMA bf16 GEMM: C[M,N] = A[M,K] * W[N,K]^T (+bias)(+resid)(relu) ----
// block = 256 threads (8 waves), block tile 128(M) x 64(N).
// Weight tile (64 rows x K) is DMA'd into LDS by the Tensor Data Mover (wave 0),
// then each wave computes a 32x32 tile: 2 A-frags (global) x 2 B-frags (LDS),
// 4 x v_wmma_f32_16x16x32_bf16 per K-step.
__global__ void __launch_bounds__(256) gemm_bf16_wmma(
    const bf16* __restrict__ A, int lda,
    const bf16* __restrict__ W, int Kdim, int N,
    const float* __restrict__ bias, const float* __restrict__ resid,
    float* __restrict__ outF, bf16* __restrict__ outB, int relu) {
  extern __shared__ __align__(16) bf16 bsh[];   // 64 * Kdim bf16
  int wave = threadIdx.x >> 5;
  int lane = threadIdx.x & 31;
  int mi = wave >> 1;            // 0..3  (M direction)
  int ni = wave & 1;             // 0..1  (N direction)
  int m0 = blockIdx.y * 128 + mi * 32;
  int nblk = blockIdx.x * 64;    // global column of LDS tile row 0
  int nl = ni * 32;              // local column base inside LDS tile
  int r  = lane & 15;
  int hi = lane >> 4;

  // ---- TDM: DMA the 64 x Kdim bf16 weight tile into LDS (wave 0 only) ----
  if (threadIdx.x < 32) {
    unsigned long long gaddr = (unsigned long long)(uintptr_t)(W + (size_t)nblk * Kdim);
    unsigned int laddr = (unsigned int)(uintptr_t)bsh;   // LDS byte offset (addr[31:0])
    u32x4 g0;
    g0[0] = 1u;                                   // count=1, user descriptor
    g0[1] = laddr;                                // lds_addr
    g0[2] = (unsigned int)gaddr;                  // global_addr[31:0]
    g0[3] = (unsigned int)((gaddr >> 32) & 0x01FFFFFFu) | (2u << 30);  // addr[56:32] | type=2
    unsigned int kd = (unsigned int)Kdim;
    i32x8 g1;
    g1[0] = (int)(1u << 16);                      // workgroup_mask=0, data_size=1 (2 bytes)
    g1[1] = (int)((kd & 0xFFFFu) << 16);          // tensor_dim0[15:0] -> bits 63:48
    g1[2] = (int)((kd >> 16) | (64u << 16));      // tensor_dim0[31:16] | tensor_dim1[15:0]=64
    g1[3] = (int)(kd << 16);                      // tensor_dim1[31:16]=0 | tile_dim0=Kdim
    g1[4] = 64;                                   // tile_dim1=64, tile_dim2=0
    g1[5] = (int)kd;                              // tensor_dim0_stride[31:0] = Kdim
    g1[6] = 0;                                    // stride0[47:32]=0, stride1[15:0]=0
    g1[7] = 0;
    i32x4 z4 = {0, 0, 0, 0};
#if defined(__clang_major__) && __clang_major__ >= 23
    i32x8 z8 = {0, 0, 0, 0, 0, 0, 0, 0};
    __builtin_amdgcn_tensor_load_to_lds(g0, g1, z4, z4, z8, 0);
#else
    __builtin_amdgcn_tensor_load_to_lds(g0, g1, z4, z4, 0);
#endif
    __builtin_amdgcn_s_wait_tensorcnt(0);
  }
  __syncthreads();

  const bf16* ap0 = A + (size_t)(m0 + r) * lda + hi * 8;
  const bf16* ap1 = ap0 + (size_t)16 * lda;
  const bf16* bp0 = bsh + (size_t)(nl + r) * Kdim + hi * 8;
  const bf16* bp1 = bp0 + (size_t)16 * Kdim;

  v8f acc00 = {}, acc01 = {}, acc10 = {}, acc11 = {};
  for (int k0 = 0; k0 < Kdim; k0 += 32) {
    FragBF a0, a1, b0, b1;
    a0.h[0] = *(const v8bf*)(ap0);  a0.h[1] = *(const v8bf*)(ap0 + 16);
    a1.h[0] = *(const v8bf*)(ap1);  a1.h[1] = *(const v8bf*)(ap1 + 16);
    b0.h[0] = *(const v8bf*)(bp0);  b0.h[1] = *(const v8bf*)(bp0 + 16);
    b1.h[0] = *(const v8bf*)(bp1);  b1.h[1] = *(const v8bf*)(bp1 + 16);
    ap0 += 32; ap1 += 32; bp0 += 32; bp1 += 32;
    __builtin_prefetch(ap0, 0, 1);      // global_prefetch_b8
    acc00 = __builtin_amdgcn_wmma_f32_16x16x32_bf16(false, a0.v, false, b0.v, (short)0, acc00, false, false);
    acc01 = __builtin_amdgcn_wmma_f32_16x16x32_bf16(false, a0.v, false, b1.v, (short)0, acc01, false, false);
    acc10 = __builtin_amdgcn_wmma_f32_16x16x32_bf16(false, a1.v, false, b0.v, (short)0, acc10, false, false);
    acc11 = __builtin_amdgcn_wmma_f32_16x16x32_bf16(false, a1.v, false, b1.v, (short)0, acc11, false, false);
  }

  // epilogue: C/D layout: VGPR i, lanes 0-15 -> M=i, lanes 16-31 -> M=i+8; N = lane&15
  int c0 = nblk + nl + r;
  int c1 = c0 + 16;
  float bv0 = bias ? bias[c0] : 0.f;
  float bv1 = bias ? bias[c1] : 0.f;
#pragma unroll
  for (int i = 0; i < 8; ++i) {
    int row0 = m0 + i + 8 * hi;
    int row1 = row0 + 16;
    float v00 = acc00[i] + bv0, v01 = acc01[i] + bv1;
    float v10 = acc10[i] + bv0, v11 = acc11[i] + bv1;
    size_t o00 = (size_t)row0 * N + c0, o01 = (size_t)row0 * N + c1;
    size_t o10 = (size_t)row1 * N + c0, o11 = (size_t)row1 * N + c1;
    if (resid) { v00 += resid[o00]; v01 += resid[o01]; v10 += resid[o10]; v11 += resid[o11]; }
    if (relu) { v00 = fmaxf(v00, 0.f); v01 = fmaxf(v01, 0.f);
                v10 = fmaxf(v10, 0.f); v11 = fmaxf(v11, 0.f); }
    if (outF) { outF[o00] = v00; outF[o01] = v01; outF[o10] = v10; outF[o11] = v11; }
    if (outB) { outB[o00] = (bf16)v00; outB[o01] = (bf16)v01;
                outB[o10] = (bf16)v10; outB[o11] = (bf16)v11; }
  }
}

// ---------------- temporal attention: one block per (e, head) ----------------
__global__ void __launch_bounds__(64) temporal_attn_kernel(
    const bf16* __restrict__ q, const bf16* __restrict__ k,
    const bf16* __restrict__ v, bf16* __restrict__ o) {
  __shared__ bf16 ks[NT * DH];
  __shared__ bf16 vs[NT * DH];
  int e = blockIdx.x >> 3;
  int h = blockIdx.x & 7;
  size_t base = ((size_t)e * NT) * ND + h * DH;
  for (int idx = threadIdx.x; idx < NT * DH; idx += 64) {
    int t = idx >> 4, d = idx & 15;
    ks[idx] = k[base + (size_t)t * ND + d];
    vs[idx] = v[base + (size_t)t * ND + d];
  }
  __syncthreads();
  int t = threadIdx.x;
  if (t < NT) {
    const bf16* qp = q + base + (size_t)t * ND;
    float qr[DH];
#pragma unroll
    for (int j = 0; j < DH; ++j) qr[j] = (float)qp[j];
    float s[NT];
    float mx = -1e30f;
    for (int u = 0; u < NT; ++u) {
      float d = 0.f;
#pragma unroll
      for (int j = 0; j < DH; ++j) d += qr[j] * (float)ks[u * DH + j];
      d *= 0.25f;                     // 1/sqrt(16)
      s[u] = d;
      mx = fmaxf(mx, d);
    }
    float sum = 0.f;
    for (int u = 0; u < NT; ++u) { s[u] = __expf(s[u] - mx); sum += s[u]; }
    float inv = 1.f / sum;
    float z[DH];
#pragma unroll
    for (int j = 0; j < DH; ++j) z[j] = 0.f;
    for (int u = 0; u < NT; ++u) {
      float a = s[u];
#pragma unroll
      for (int j = 0; j < DH; ++j) z[j] += a * (float)vs[u * DH + j];
    }
    bf16* op = o + base + (size_t)t * ND;
#pragma unroll
    for (int j = 0; j < DH; ++j) op[j] = (bf16)(z[j] * inv);
  }
}

// ---------------- spatial neighbor attention: one block per node e ----------------
__global__ void __launch_bounds__(256) spatial_attn_kernel(
    const bf16* __restrict__ q, const bf16* __restrict__ k0,
    const bf16* __restrict__ v0, const int* __restrict__ nbr,
    bf16* __restrict__ o) {
  __shared__ bf16 kn[KNBR * ND];
  __shared__ bf16 vn[KNBR * ND];
  __shared__ int nb[KNBR];
  int e = blockIdx.x;
  if (threadIdx.x < KNBR) nb[threadIdx.x] = nbr[e * KNBR + threadIdx.x];
  __syncthreads();
  for (int idx = threadIdx.x; idx < KNBR * ND; idx += 256) {
    int kk = idx >> 7, d = idx & 127;
    kn[idx] = k0[(size_t)nb[kk] * ND + d];
    vn[idx] = v0[(size_t)nb[kk] * ND + d];
  }
  __syncthreads();
  for (int pi = threadIdx.x; pi < NT * NH; pi += 256) {
    int t = pi >> 3, h = pi & 7;
    const bf16* qp = q + ((size_t)e * NT + t) * ND + h * DH;
    float qr[DH];
#pragma unroll
    for (int j = 0; j < DH; ++j) qr[j] = (float)qp[j];
    float s[KNBR];
    float mx = -1e30f;
#pragma unroll
    for (int kk = 0; kk < KNBR; ++kk) {
      float d = 0.f;
#pragma unroll
      for (int j = 0; j < DH; ++j) d += qr[j] * (float)kn[kk * ND + h * DH + j];
      d *= 0.25f;
      s[kk] = d;
      mx = fmaxf(mx, d);
    }
    float sum = 0.f;
#pragma unroll
    for (int kk = 0; kk < KNBR; ++kk) { s[kk] = __expf(s[kk] - mx); sum += s[kk]; }
    float inv = 1.f / sum;
    float z[DH];
#pragma unroll
    for (int j = 0; j < DH; ++j) z[j] = 0.f;
#pragma unroll
    for (int kk = 0; kk < KNBR; ++kk) {
      float a = s[kk];
#pragma unroll
      for (int j = 0; j < DH; ++j) z[j] += a * (float)vn[kk * ND + h * DH + j];
    }
    bf16* op = o + ((size_t)e * NT + t) * ND + h * DH;
#pragma unroll
    for (int j = 0; j < DH; ++j) op[j] = (bf16)(z[j] * inv);
  }
}

// ---------------- LayerNorm: one wave32 per row of 128 ----------------
__global__ void __launch_bounds__(256) ln_kernel(
    const float* __restrict__ in, const float* __restrict__ g,
    const float* __restrict__ b, float* __restrict__ outF,
    bf16* __restrict__ outB) {
  size_t row = (size_t)blockIdx.x * 8 + (threadIdx.x >> 5);
  int lane = threadIdx.x & 31;
  const float* p = in + row * ND + lane * 4;
  float x0 = p[0], x1 = p[1], x2 = p[2], x3 = p[3];
  float s = x0 + x1 + x2 + x3;
#pragma unroll
  for (int m = 16; m >= 1; m >>= 1) s += __shfl_xor(s, m, 32);
  float mean = s * (1.f / 128.f);
  float d0 = x0 - mean, d1 = x1 - mean, d2 = x2 - mean, d3 = x3 - mean;
  float vv = d0 * d0 + d1 * d1 + d2 * d2 + d3 * d3;
#pragma unroll
  for (int m = 16; m >= 1; m >>= 1) vv += __shfl_xor(vv, m, 32);
  float rinv = rsqrtf(vv * (1.f / 128.f) + 1e-5f);
  int c = lane * 4;
  float y0 = d0 * rinv * g[c + 0] + b[c + 0];
  float y1 = d1 * rinv * g[c + 1] + b[c + 1];
  float y2 = d2 * rinv * g[c + 2] + b[c + 2];
  float y3 = d3 * rinv * g[c + 3] + b[c + 3];
  float* q = outF + row * ND + c;
  q[0] = y0; q[1] = y1; q[2] = y2; q[3] = y3;
  bf16* qb = outB + row * ND + c;
  qb[0] = (bf16)y0; qb[1] = (bf16)y1; qb[2] = (bf16)y2; qb[3] = (bf16)y3;
}

// ---------------- head: softplus(h @ w_head^T + b) ----------------
__global__ void __launch_bounds__(256) head_kernel(
    const float* __restrict__ h, const float* __restrict__ w,
    const float* __restrict__ b, float* __restrict__ out) {
  size_t row = (size_t)blockIdx.x * 256 + threadIdx.x;
  const float* p = h + row * ND;
  float acc = b[0];
#pragma unroll 8
  for (int d = 0; d < ND; ++d) acc += p[d] * w[d];
  out[row] = (acc > 20.f) ? acc : log1pf(__expf(acc));
}

// ---------------- host side ----------------
struct SB { const float *b1,*b2,*bo,*ln1b,*ln1g,*ln2b,*ln2g,*w1,*w2,*wk,*wo,*wq,*wv; };
struct TB { const float *b1,*b2,*bk,*bo,*bq,*bv,*ln1b,*ln1g,*ln2b,*ln2g,*w1,*w2,*wk,*wo,*wq,*wv; };

extern "C" void kernel_launch(void* const* d_in, const int* in_sizes, int n_in,
                              void* d_out, int out_size, void* d_ws, size_t ws_size,
                              hipStream_t stream) {
  (void)in_sizes; (void)n_in; (void)out_size; (void)ws_size;
  auto f = [&](int i) { return (const float*)d_in[i]; };

  const float* in_x     = f(0);
  const float* in_hours = f(1);
  /* dows (d_in[2]) unused by the reference */
  const float* in_U     = f(3);
  const int*   in_nbr   = (const int*)d_in[4];

  // params flattened after the 5 data inputs, sorted-key tree order:
  // b_head, b_in, layers[0]{s{...},t{...}}, layers[1]{...}, t2v_bl, t2v_bp,
  // t2v_wl, t2v_wp, w_head, w_in, w_pe
  int pi = 5;
  const float* p_b_head = f(pi++);
  const float* p_b_in   = f(pi++);
  SB S[NL]; TB Tt[NL];
  for (int l = 0; l < NL; ++l) {
    SB& s = S[l];
    s.b1 = f(pi++); s.b2 = f(pi++); s.bo = f(pi++);
    s.ln1b = f(pi++); s.ln1g = f(pi++); s.ln2b = f(pi++); s.ln2g = f(pi++);
    s.w1 = f(pi++); s.w2 = f(pi++); s.wk = f(pi++); s.wo = f(pi++); s.wq = f(pi++); s.wv = f(pi++);
    TB& t = Tt[l];
    t.b1 = f(pi++); t.b2 = f(pi++); t.bk = f(pi++); t.bo = f(pi++); t.bq = f(pi++); t.bv = f(pi++);
    t.ln1b = f(pi++); t.ln1g = f(pi++); t.ln2b = f(pi++); t.ln2g = f(pi++);
    t.w1 = f(pi++); t.w2 = f(pi++); t.wk = f(pi++); t.wo = f(pi++); t.wq = f(pi++); t.wv = f(pi++);
  }
  const float* p_t2v_bl = f(pi++);
  const float* p_t2v_bp = f(pi++);
  const float* p_t2v_wl = f(pi++);
  const float* p_t2v_wp = f(pi++);
  const float* p_w_head = f(pi++);
  const float* p_w_in   = f(pi++);
  const float* p_w_pe   = f(pi++);

  // ---- workspace layout ----
  const size_t HFB = MROWS * ND * sizeof(float);       // 100663296
  char* base = (char*)d_ws;
  float* hF = (float*)(base);
  float* P  = (float*)(base + HFB);
  float* Y  = (float*)(base + 2 * HFB);
  bf16*  hB = (bf16*) (base + 3 * HFB);
  bf16*  qB = (bf16*) (base + 3 * HFB + HFB / 2);
  bf16*  yB = (bf16*) (base + 4 * HFB);
  char*  SH = base + 4 * HFB + HFB / 2;                // 2*HFB shared region
  bf16*  kB = (bf16*)SH;
  bf16*  vB = (bf16*)(SH + HFB / 2);
  bf16*  oB = (bf16*)(SH + HFB);
  bf16*  ffB = (bf16*)SH;                              // overlaps k/v (dead by then)
  char*  tail = SH + 2 * HFB;
  bf16*  k0B = (bf16*)tail;
  bf16*  v0B = (bf16*)(tail + (size_t)NE * ND * 2);
  char*  wcur = tail + 2 * (size_t)NE * ND * 2;

  auto cvt = [&](const float* src, size_t n) -> const bf16* {
    bf16* dst = (bf16*)wcur;
    wcur += ((n * 2 + 255) / 256) * 256;
    f2b_kernel<<<(unsigned)((n + 255) / 256), 256, 0, stream>>>(src, dst, (int)n);
    return dst;
  };

  // convert all GEMM weights to bf16 (tiny: ~1.3 MB)
  const bf16 *twq[NL], *twk[NL], *twv[NL], *two[NL], *tw1[NL], *tw2[NL];
  const bf16 *swq[NL], *swk[NL], *swv[NL], *swo[NL], *sw1[NL], *sw2[NL];
  for (int l = 0; l < NL; ++l) {
    twq[l] = cvt(Tt[l].wq, ND * ND);   twk[l] = cvt(Tt[l].wk, ND * ND);
    twv[l] = cvt(Tt[l].wv, ND * ND);   two[l] = cvt(Tt[l].wo, ND * ND);
    tw1[l] = cvt(Tt[l].w1, DFF_T * ND); tw2[l] = cvt(Tt[l].w2, ND * DFF_T);
    swq[l] = cvt(S[l].wq, ND * ND);    swk[l] = cvt(S[l].wk, ND * ND);
    swv[l] = cvt(S[l].wv, ND * ND);    swo[l] = cvt(S[l].wo, ND * ND);
    sw1[l] = cvt(S[l].w1, DFF_S * ND); sw2[l] = cvt(S[l].w2, ND * DFF_S);
  }

  auto gemm = [&](const bf16* A, int lda, const bf16* W, int K, int N, size_t Mr,
                  const float* bias, const float* resid, float* oF, bf16* oBt, int relu) {
    dim3 g((unsigned)(N / 64), (unsigned)(Mr / 128));
    size_t smem = (size_t)64 * K * sizeof(bf16);       // LDS weight tile
    gemm_bf16_wmma<<<g, 256, smem, stream>>>(A, lda, W, K, N, bias, resid, oF, oBt, relu);
  };

  // ---- embedding ----
  embed_kernel<<<(unsigned)(MROWS * ND / 256), 256, 0, stream>>>(
      in_x, in_hours, in_U, p_t2v_wl, p_t2v_bl, p_t2v_wp, p_t2v_bp,
      p_w_in, p_b_in, p_w_pe, hF, hB);

  const unsigned LN_G = (unsigned)(MROWS / 8);

  for (int l = 0; l < NL; ++l) {
    // ===== temporal block =====
    gemm(hB, ND, twq[l], ND, ND, MROWS, Tt[l].bq, nullptr, nullptr, qB, 0);
    gemm(hB, ND, twk[l], ND, ND, MROWS, Tt[l].bk, nullptr, nullptr, kB, 0);
    gemm(hB, ND, twv[l], ND, ND, MROWS, Tt[l].bv, nullptr, nullptr, vB, 0);
    temporal_attn_kernel<<<(unsigned)(NE * NH), 64, 0, stream>>>(qB, kB, vB, oB);
    gemm(oB, ND, two[l], ND, ND, MROWS, Tt[l].bo, hF, P, nullptr, 0);
    ln_kernel<<<LN_G, 256, 0, stream>>>(P, Tt[l].ln1g, Tt[l].ln1b, Y, yB);
    gemm(yB, ND, tw1[l], ND, DFF_T, MROWS, Tt[l].b1, nullptr, nullptr, ffB, 1);
    gemm(ffB, DFF_T, tw2[l], DFF_T, ND, MROWS, Tt[l].b2, Y, hF, nullptr, 0);
    ln_kernel<<<LN_G, 256, 0, stream>>>(hF, Tt[l].ln2g, Tt[l].ln2b, hF, hB);

    // ===== spatial block =====
    gemm(hB, ND, swq[l], ND, ND, MROWS, nullptr, nullptr, nullptr, qB, 0);
    // K/V from t=0 slice: row e is at hB + e*(NT*ND)
    gemm(hB, NT * ND, swk[l], ND, ND, NE, nullptr, nullptr, nullptr, k0B, 0);
    gemm(hB, NT * ND, swv[l], ND, ND, NE, nullptr, nullptr, nullptr, v0B, 0);
    spatial_attn_kernel<<<(unsigned)NE, 256, 0, stream>>>(qB, k0B, v0B, in_nbr, oB);
    gemm(oB, ND, swo[l], ND, ND, MROWS, S[l].bo, hF, P, nullptr, 0);
    ln_kernel<<<LN_G, 256, 0, stream>>>(P, S[l].ln1g, S[l].ln1b, Y, yB);
    gemm(yB, ND, sw1[l], ND, DFF_S, MROWS, S[l].b1, nullptr, nullptr, ffB, 1);
    gemm(ffB, DFF_S, sw2[l], DFF_S, ND, MROWS, S[l].b2, Y, hF, nullptr, 0);
    ln_kernel<<<LN_G, 256, 0, stream>>>(hF, S[l].ln2g, S[l].ln2b, hF, hB);
  }

  // ---- head ----
  head_kernel<<<(unsigned)(MROWS / 256), 256, 0, stream>>>(hF, p_w_head, p_b_head,
                                                           (float*)d_out);
}
